// MultiHeadAttention_65807488909915
// MI455X (gfx1250) — matrix operand order repaired and checked by reading, not compiled, not run on hardware
//
#include <hip/hip_runtime.h>

typedef __bf16 bf16;
typedef __attribute__((ext_vector_type(16))) __bf16 v16bf;
typedef __attribute__((ext_vector_type(8)))  float v8f;
typedef __attribute__((ext_vector_type(4)))  unsigned int u32x4;

union AFrag { v16bf v; u32x4 u[2]; };

#define N_HEADS 16
#define HEAD_DIM 64
#define N_EMBD 1024
#define NB 2
#define SS 2048
#define M_ROWS (NB * SS)      /* 4096 */
#define QKV_N (3 * N_EMBD)    /* 3072 */

static __device__ inline u32x4 ld16(const bf16* p) { return *(const u32x4*)p; }

static __device__ inline v8f wmma_bf16(const AFrag& a, const AFrag& b, v8f c) {
  return __builtin_amdgcn_wmma_f32_16x16x32_bf16(false, a.v, false, b.v,
                                                 (short)0, c, false, false);
}

// ---------------- conversion / transpose prepasses ----------------

__global__ void cvt_kernel(const float* __restrict__ in, bf16* __restrict__ out, int n) {
  int i = blockIdx.x * blockDim.x + threadIdx.x;
  if (i < n) out[i] = (bf16)in[i];
}

// in: K x N row-major fp32 ; out: N x K row-major bf16 (i.e. transposed)
__global__ void transpose_cvt_kernel(const float* __restrict__ in, bf16* __restrict__ out,
                                     int K, int N) {
  int i = blockIdx.x * blockDim.x + threadIdx.x;
  if (i < K * N) {
    int nn = i / K, kk = i - nn * K;
    out[i] = (bf16)in[(size_t)kk * N + nn];
  }
}

// ---------------- 32x64 wave GEMM core, register double-buffered ----------------

template <int MT>
static __device__ inline void wg_loadA(AFrag (&a)[MT], const bf16* const (&arow)[MT],
                                       int k, int khalf) {
#pragma unroll
  for (int i = 0; i < MT; ++i) {
    a[i].u[0] = ld16(arow[i] + k + 2 * khalf);
    a[i].u[1] = ld16(arow[i] + k + 8 + 2 * khalf);
  }
}

template <int NT>
static __device__ inline void wg_loadB(AFrag (&b)[NT], const bf16* const (&brow)[NT], int k) {
#pragma unroll
  for (int n = 0; n < NT; ++n) {
    b[n].u[0] = ld16(brow[n] + k);
    b[n].u[1] = ld16(brow[n] + k + 8);
  }
}

template <int MT, int NT>
static __device__ inline void wg_mma(v8f (&acc)[MT][NT], const AFrag (&a)[MT],
                                     const AFrag (&b)[NT]) {
#pragma unroll
  for (int i = 0; i < MT; ++i)
#pragma unroll
    for (int n = 0; n < NT; ++n)
      acc[i][n] = wmma_bf16(a[i], b[n], acc[i][n]);
}

// C(MT*16 x NT*16) = A[row0.., 0:K] (bf16 row-major lda) @ Bt[col0.., 0:K]^T (bf16 ldb)
// K must be a multiple of 64.
template <int MT, int NT>
static __device__ inline void wave_gemm(const bf16* __restrict__ A, int lda,
                                        const bf16* __restrict__ Bt, int ldb,
                                        int row0, int col0, int K, v8f (&acc)[MT][NT]) {
  const int lane  = threadIdx.x & 31;
  const int m     = lane & 15;
  const int khalf = (lane >> 4) << 3;   // 0 / 8
  const int kb16  = (lane >> 4) << 4;   // 0 / 16

  const bf16* arow[MT];
#pragma unroll
  for (int i = 0; i < MT; ++i) arow[i] = A + (size_t)(row0 + i * 16 + m) * lda;
  const bf16* brow[NT];
#pragma unroll
  for (int n = 0; n < NT; ++n) brow[n] = Bt + (size_t)(col0 + n * 16 + m) * ldb + kb16;

  AFrag a0[MT], a1[MT], b0[NT], b1[NT];
  wg_loadA(a0, arow, 0, khalf);
  wg_loadB(b0, brow, 0);
  for (int k = 0; k < K; k += 64) {
    wg_loadA(a1, arow, k + 32, khalf);   // prefetch next 32-step
    wg_loadB(b1, brow, k + 32);
    wg_mma(acc, a0, b0);                 // compute current (loaded last iter)
    if (k + 64 < K) {
      wg_loadA(a0, arow, k + 64, khalf);
      wg_loadB(b0, brow, k + 64);
    }
    wg_mma(acc, a1, b1);
  }
}

// ---------------- QKV projection ----------------
__global__ __launch_bounds__(256) void qkv_kernel(const bf16* __restrict__ xb,
                                                  const bf16* __restrict__ wat,
                                                  const float* __restrict__ b_attn,
                                                  bf16* __restrict__ Qo,
                                                  bf16* __restrict__ Ko,
                                                  bf16* __restrict__ Vto) {
  const int wid = blockIdx.x * 8 + (threadIdx.x >> 5);
  const int ntn = QKV_N / 64;               // 48
  const int tm = wid / ntn, tn = wid - tm * ntn;
  const int row0 = tm * 32, col0 = tn * 64;
  v8f acc[2][4] = {{{}, {}, {}, {}}, {{}, {}, {}, {}}};
  wave_gemm<2, 4>(xb, N_EMBD, wat, N_EMBD, row0, col0, N_EMBD, acc);

  const int lane = threadIdx.x & 31;
  const int m = lane & 15, half = lane >> 4;
#pragma unroll
  for (int n = 0; n < 4; ++n) {
    const int col = col0 + n * 16 + m;
    const float bias = b_attn[col];
#pragma unroll
    for (int mt = 0; mt < 2; ++mt) {
#pragma unroll
      for (int r = 0; r < 8; ++r) {
        const int row = row0 + mt * 16 + r + half * 8;
        const float v = acc[mt][n][r] + bias;
        const int bidx = row >> 11, s = row & (SS - 1);
        if (col < N_EMBD) {                       // Q -> [b,h,s,d]
          const int h = col >> 6, d = col & 63;
          Qo[((size_t)(bidx * N_HEADS + h) * SS + s) * HEAD_DIM + d] = (bf16)v;
        } else if (col < 2 * N_EMBD) {            // K -> [b,h,s,d]
          const int e = col - N_EMBD, h = e >> 6, d = e & 63;
          Ko[((size_t)(bidx * N_HEADS + h) * SS + s) * HEAD_DIM + d] = (bf16)v;
        } else {                                  // V -> transposed [b,h,d,s]
          const int e = col - 2 * N_EMBD, h = e >> 6, d = e & 63;
          Vto[((size_t)(bidx * N_HEADS + h) * HEAD_DIM + d) * SS + s] = (bf16)v;
        }
      }
    }
  }
}

// ---------------- fused causal flash attention ----------------
struct KF { AFrag f[4]; };   // [s0 kk0, s0 kk1, s1 kk0, s1 kk1]

__global__ __launch_bounds__(32) void flash_kernel(const bf16* __restrict__ Q,
                                                   const bf16* __restrict__ Kmat,
                                                   const bf16* __restrict__ Vt,
                                                   bf16* __restrict__ Ob) {
  const int lane = threadIdx.x & 31;
  const int m = lane & 15, half = lane >> 4;
  const int khalf = half * 8, kb16 = half * 16;
  const int bh = blockIdx.x >> 7;             // 0..31 = b*16+h
  const int q0 = (blockIdx.x & 127) << 4;     // query block start

  const bf16* Qh = Q    + (size_t)bh * SS * HEAD_DIM;
  const bf16* Kh = Kmat + (size_t)bh * SS * HEAD_DIM;
  const bf16* Vh = Vt   + (size_t)bh * HEAD_DIM * SS;

  AFrag qf[2];
#pragma unroll
  for (int kk = 0; kk < 2; ++kk) {
    const bf16* p = Qh + (size_t)(q0 + m) * HEAD_DIM + kk * 32 + kb16;
    qf[kk].u[0] = ld16(p);
    qf[kk].u[1] = ld16(p + 8);
  }

  v8f o0 = {}, o1 = {}, o2 = {}, o3 = {};
  v8f mrow, lrow;
#pragma unroll
  for (int r = 0; r < 8; ++r) { mrow[r] = -1e30f; lrow[r] = 0.0f; }

  __shared__ __align__(16) bf16 Pl[16 * 32];

  auto load_k = [&](KF& kf, int tb) {
#pragma unroll
    for (int kk = 0; kk < 2; ++kk) {
      const bf16* p0 = Kh + (size_t)(tb + m) * HEAD_DIM + kk * 32 + kb16;
      kf.f[kk].u[0] = ld16(p0);     kf.f[kk].u[1] = ld16(p0 + 8);
      const bf16* p1 = Kh + (size_t)(tb + 16 + m) * HEAD_DIM + kk * 32 + kb16;
      kf.f[2 + kk].u[0] = ld16(p1); kf.f[2 + kk].u[1] = ld16(p1 + 8);
    }
  };

  auto tile_step = [&](int tb, KF& kc, KF& kn, bool pre) {
    // V fragments first: independent of softmax, latency hidden under QK+softmax
    AFrag vf[4];
#pragma unroll
    for (int n = 0; n < 4; ++n) {
      const bf16* pv = Vh + (size_t)(n * 16 + m) * SS + tb + kb16;
      vf[n].u[0] = ld16(pv);
      vf[n].u[1] = ld16(pv + 8);
    }

    v8f s0 = {}, s1 = {};
    s0 = wmma_bf16(qf[0], kc.f[0], s0);
    s0 = wmma_bf16(qf[1], kc.f[1], s0);
    s1 = wmma_bf16(qf[0], kc.f[2], s1);
    s1 = wmma_bf16(qf[1], kc.f[3], s1);

    if (pre) load_k(kn, tb + 32);        // prefetch next tile's K fragments

    s0 *= 0.125f;  // 1/sqrt(64)
    s1 *= 0.125f;

    if (tb + 31 > q0) {                  // tile touches the diagonal
#pragma unroll
      for (int r = 0; r < 8; ++r) {
        const int q = q0 + r + half * 8;
        if (tb + m > q)      s0[r] = -1e30f;
        if (tb + 16 + m > q) s1[r] = -1e30f;
      }
    }

    v8f alpha;
#pragma unroll
    for (int r = 0; r < 8; ++r) {
      float mx = fmaxf(s0[r], s1[r]);
      mx = fmaxf(mx, __shfl_xor(mx, 1, 32));
      mx = fmaxf(mx, __shfl_xor(mx, 2, 32));
      mx = fmaxf(mx, __shfl_xor(mx, 4, 32));
      mx = fmaxf(mx, __shfl_xor(mx, 8, 32));
      const float mn = fmaxf(mrow[r], mx);
      const float al = __expf(mrow[r] - mn);
      const float p0 = __expf(s0[r] - mn);
      const float p1 = __expf(s1[r] - mn);
      float rs = p0 + p1;
      rs += __shfl_xor(rs, 1, 32);
      rs += __shfl_xor(rs, 2, 32);
      rs += __shfl_xor(rs, 4, 32);
      rs += __shfl_xor(rs, 8, 32);
      lrow[r] = lrow[r] * al + rs;
      mrow[r] = mn;
      alpha[r] = al;
      Pl[(r + half * 8) * 32 + m]      = (bf16)p0;
      Pl[(r + half * 8) * 32 + 16 + m] = (bf16)p1;
    }
#pragma unroll
    for (int r = 0; r < 8; ++r) {
      o0[r] *= alpha[r]; o1[r] *= alpha[r]; o2[r] *= alpha[r]; o3[r] *= alpha[r];
    }

    // P tile: C-layout -> A-layout via LDS (same-wave LDS is in-order)
    AFrag pf;
    pf.u[0] = *(const u32x4*)&Pl[m * 32 + kb16];
    pf.u[1] = *(const u32x4*)&Pl[m * 32 + 8 + kb16];

    o0 = wmma_bf16(pf, vf[0], o0);
    o1 = wmma_bf16(pf, vf[1], o1);
    o2 = wmma_bf16(pf, vf[2], o2);
    o3 = wmma_bf16(pf, vf[3], o3);
  };

  const int tmax = (q0 + 15) >> 5;
  KF kA, kB;
  load_k(kA, 0);
  for (int t = 0; t <= tmax; t += 2) {   // 2x unrolled for K-fragment ping-pong
    tile_step(t << 5, kA, kB, t < tmax);
    if (t + 1 <= tmax) tile_step((t + 1) << 5, kB, kA, t + 1 < tmax);
  }

  const int b = bh >> 4, h = bh & 15;
#pragma unroll
  for (int r = 0; r < 8; ++r) {
    const float inv = 1.0f / lrow[r];
    const int s = q0 + r + half * 8;
    const size_t off = (size_t)(b * SS + s) * N_EMBD + h * HEAD_DIM + m;
    Ob[off +  0] = (bf16)(o0[r] * inv);
    Ob[off + 16] = (bf16)(o1[r] * inv);
    Ob[off + 32] = (bf16)(o2[r] * inv);
    Ob[off + 48] = (bf16)(o3[r] * inv);
  }
}

// ---------------- output projection ----------------
__global__ __launch_bounds__(256) void proj_kernel(const bf16* __restrict__ ab,
                                                   const bf16* __restrict__ wpt,
                                                   const float* __restrict__ b_proj,
                                                   float* __restrict__ out) {
  const int wid = blockIdx.x * 8 + (threadIdx.x >> 5);
  const int ntn = N_EMBD / 64;              // 16
  const int tm = wid / ntn, tn = wid - tm * ntn;
  const int row0 = tm * 32, col0 = tn * 64;
  v8f acc[2][4] = {{{}, {}, {}, {}}, {{}, {}, {}, {}}};
  wave_gemm<2, 4>(ab, N_EMBD, wpt, N_EMBD, row0, col0, N_EMBD, acc);

  const int lane = threadIdx.x & 31;
  const int m = lane & 15, half = lane >> 4;
#pragma unroll
  for (int n = 0; n < 4; ++n) {
    const int col = col0 + n * 16 + m;
    const float bias = b_proj[col];
#pragma unroll
    for (int mt = 0; mt < 2; ++mt) {
#pragma unroll
      for (int r = 0; r < 8; ++r) {
        const int row = row0 + mt * 16 + r + half * 8;
        out[(size_t)row * N_EMBD + col] = acc[mt][n][r] + bias;
      }
    }
  }
}

// ---------------- launch ----------------
extern "C" void kernel_launch(void* const* d_in, const int* in_sizes, int n_in,
                              void* d_out, int out_size, void* d_ws, size_t ws_size,
                              hipStream_t stream) {
  const float* x      = (const float*)d_in[0];
  const float* W_attn = (const float*)d_in[1];
  const float* b_attn = (const float*)d_in[2];
  const float* W_proj = (const float*)d_in[3];
  const float* b_proj = (const float*)d_in[4];
  float* out = (float*)d_out;

  char* ws = (char*)d_ws;
  bf16* xb  = (bf16*)(ws + 0);         // 4096x1024            (8 MiB)
  bf16* wat = (bf16*)(ws + 8388608);   // 3072x1024 (W_attn^T) (6 MiB)
  bf16* wpt = (bf16*)(ws + 14680064);  // 1024x1024 (W_proj^T) (2 MiB)
  bf16* Qo  = (bf16*)(ws + 16777216);  // [b,h,s,64]           (8 MiB)
  bf16* Ko  = (bf16*)(ws + 25165824);  // [b,h,s,64]           (8 MiB)
  bf16* Vto = (bf16*)(ws + 33554432);  // [b,h,64,s]           (8 MiB)
  bf16* Ab  = (bf16*)(ws + 41943040);  // 4096x1024 attn out   (8 MiB)

  cvt_kernel<<<(M_ROWS * N_EMBD) / 256, 256, 0, stream>>>(x, xb, M_ROWS * N_EMBD);
  transpose_cvt_kernel<<<(N_EMBD * QKV_N) / 256, 256, 0, stream>>>(W_attn, wat, N_EMBD, QKV_N);
  transpose_cvt_kernel<<<(N_EMBD * N_EMBD) / 256, 256, 0, stream>>>(W_proj, wpt, N_EMBD, N_EMBD);

  // 4096/32 * 3072/64 = 6144 waves, 8 waves per block
  qkv_kernel<<<768, 256, 0, stream>>>(xb, wat, b_attn, Qo, Ko, Vto);

  // one wave per 16-query block per (b,h): 2*16*128 = 4096 blocks
  flash_kernel<<<4096, 32, 0, stream>>>(Qo, Ko, Vto, Ab);

  // 4096/32 * 1024/64 = 2048 waves, 8 per block
  proj_kernel<<<256, 256, 0, stream>>>(Ab, wpt, b_proj, out);
}
